// RNN_62508954026437
// MI455X (gfx1250) — matrix operand order repaired and checked by reading, not compiled, not run on hardware
//
#include <hip/hip_runtime.h>
#include <hip/hip_bf16.h>

// ---------------- problem constants ----------------
#define BB   2048          // batch
#define IN_  4
#define HH   1024          // hidden
#define LL   4             // layers
#define OUTN 256           // decoder out
#define GG   (3 * HH)      // gate rows per layer

typedef unsigned short u16;
typedef __attribute__((ext_vector_type(16))) __bf16 v16bf;
typedef __attribute__((ext_vector_type(8)))  float  v8f;

// async global->LDS staging path (gfx1250), guarded so absence falls back.
// Builtin signature (from clang diagnostic): arg0 = AS1 (device) pointer to
// int __vector_size__(16), arg1 = AS3 (LDS) pointer, arg2/arg3 = imm ints.
#if defined(__HIP_DEVICE_COMPILE__) && defined(__gfx1250__) && \
    __has_builtin(__builtin_amdgcn_global_load_async_to_lds_b128) && \
    __has_builtin(__builtin_amdgcn_s_wait_asynccnt)
#define HAVE_ASYNC_LDS 1
typedef int v4i_g __attribute__((__vector_size__(4 * sizeof(int))));
typedef __attribute__((address_space(1))) v4i_g* as1_v4i;
typedef __attribute__((address_space(3))) v4i_g* as3_v4i;
#endif

// ---------------- helpers ----------------
__device__ __forceinline__ u16 f2bf(float f) {
    union { float f; unsigned u; } x; x.f = f;
    unsigned r = x.u + 0x7FFFu + ((x.u >> 16) & 1u);   // round-to-nearest-even
    return (u16)(r >> 16);
}

__device__ __forceinline__ float sigm(float x) {
    return 1.0f / (1.0f + __expf(-x));
}

union FragU { uint4 u[2]; v16bf v; };

// A fragment: 16x32 bf16, M row = lane&15, per ISA layout:
//   lanes 0-15:  K = [k0..k0+7] in v0..3,  [k0+16..k0+23] in v4..7
//   lanes 16-31: K = [k0+8..k0+15],        [k0+24..k0+31]
__device__ __forceinline__ v16bf loadA(const u16* __restrict__ base,
                                       int row, int k0, int half) {
    const u16* p = base + (size_t)row * HH + k0 + half * 8;
    FragU f;
    f.u[0] = *(const uint4*)(p);
    f.u[1] = *(const uint4*)(p + 16);
    return f.v;
}

#define WMMA_BF16(A, B, C) \
    __builtin_amdgcn_wmma_f32_16x16x32_bf16(false, (A), false, (B), (short)0, (C), false, false)

// ---------------- kernels ----------------
__global__ void cvt_f32_bf16(const float* __restrict__ src, u16* __restrict__ dst, int n) {
    int i = blockIdx.x * blockDim.x + threadIdx.x;
    int stride = gridDim.x * blockDim.x;
    for (; i < n; i += stride) dst[i] = f2bf(src[i]);
}

// x0 = input @ i2h_w.T + i2h_b   (K=4, trivially memory bound)
__global__ void i2h_proj(const float* __restrict__ in, const float* __restrict__ w,
                         const float* __restrict__ b, u16* __restrict__ xo) {
    int i = blockIdx.x * blockDim.x + threadIdx.x;
    if (i >= BB * HH) return;
    int bi = i >> 10, h = i & (HH - 1);
    const float* ir = in + bi * IN_;
    const float* wr = w + h * IN_;
    float s = b[h] + ir[0] * wr[0] + ir[1] * wr[1] + ir[2] * wr[2] + ir[3] * wr[3];
    xo[i] = f2bf(s);
}

// Fused GRU layer.
// Block: 256 threads = 8 waves, covers 256 rows x 16 cols of h_new.
// Wave:  32 rows x 16 cols (M blocked by 2), 12 WMMA accumulators.
// Weights (6 fragments: Wr,Wz,Wn,Ur,Uz,Un for this N tile) staged in LDS in
// K-slabs of 64, double-buffered (24 KB), shared by all 8 waves.
// One stage = 768 b128 chunks = exactly 3 per thread (uniform, no divergence).
#define KSLAB  64
#define FRAG_H (16 * KSLAB)        // halves per fragment slab (16 rows x 64 K)
#define BUF_H  (6 * FRAG_H)        // halves per slab buffer (12 KB)

__global__ void __launch_bounds__(256)
gru_layer(const u16* __restrict__ xb,      // [B,H] bf16 layer input
          const u16* __restrict__ hb,      // [B,H] bf16 old hidden
          const u16* __restrict__ wih,     // [3H,H] bf16
          const u16* __restrict__ whh,     // [3H,H] bf16
          const float* __restrict__ bih,   // [3H]
          const float* __restrict__ bhh,   // [3H]
          const float* __restrict__ hin,   // [B,H] fp32 old hidden
          float* __restrict__ hout,        // [B,H] fp32 new hidden (d_out)
          u16* __restrict__ xnext)         // [B,H] bf16 next-layer input
{
    __shared__ __align__(16) u16 smem[2 * BUF_H];   // 24 KB

    const int t    = threadIdx.x;
    const int lane = t & 31;
    const int wave = t >> 5;
    const int half = lane >> 4;
    const int lcol = lane & 15;
    const int n0   = blockIdx.x * 16;               // N tile within H
    const int m0   = (blockIdx.y * 8 + wave) * 32;  // 32-row M tile within B
    const int j    = n0 + lcol;                     // output column (per lane)

    // B fragment from LDS slab: frag rows stored as 64 contiguous bf16.
    auto ldsB = [&](const u16* s, int frag, int sub) -> v16bf {
        const u16* p = s + frag * FRAG_H + lcol * KSLAB + sub * 32 + half * 8;
        FragU f;
        f.u[0] = *(const uint4*)(p);
        f.u[1] = *(const uint4*)(p + 16);
        return f.v;
    };

    // cooperative stage of 6 weight fragments (K-slab k0) into buffer p:
    // 768 b128 chunks -> 3 per thread, fully uniform
    auto stage = [&](int k0, int p) {
#pragma unroll
        for (int rr = 0; rr < 3; ++rr) {
            int i = t + rr * 256;            // 0..767
            int frag = i >> 7;               // 6 frags x 128 chunks
            int rem  = i & 127;
            int row  = rem >> 3;             // 16 rows
            int ch   = rem & 7;              // 8 chunks of 8 halves
            const u16* src = (frag < 3) ? wih : whh;
            int gate = (frag < 3) ? frag : frag - 3;
            const u16* gp = src + (size_t)(gate * HH + n0 + row) * HH + k0 + ch * 8;
            u16* lp = &smem[p * BUF_H + frag * FRAG_H + row * KSLAB + ch * 8];
#if defined(HAVE_ASYNC_LDS)
            // async DMA copy, no VGPR round-trip; tracked by ASYNCcnt
            __builtin_amdgcn_global_load_async_to_lds_b128(
                (as1_v4i)(uintptr_t)gp, (as3_v4i)(uintptr_t)lp, 0, 0);
#else
            *(uint4*)lp = *(const uint4*)gp;
#endif
        }
    };

    auto stage_fence = [&]() {
#if defined(HAVE_ASYNC_LDS)
        __builtin_amdgcn_s_wait_asynccnt(0);
#endif
        __syncthreads();
    };

    v8f air[2] = {{}, {}}, aiz[2] = {{}, {}}, ain[2] = {{}, {}};
    v8f ahr[2] = {{}, {}}, ahz[2] = {{}, {}}, ahn[2] = {{}, {}};

    // compute one K-slab (two 32-wide WMMA sub-steps) from LDS buffer sb
    auto compute = [&](int k0, const u16* sb) {
#pragma unroll
        for (int sub = 0; sub < 2; ++sub) {
            int kk = k0 + sub * 32;
            v16bf ax0 = loadA(xb, m0 + lcol,      kk, half);
            v16bf ax1 = loadA(xb, m0 + 16 + lcol, kk, half);
            v16bf ah0 = loadA(hb, m0 + lcol,      kk, half);
            v16bf ah1 = loadA(hb, m0 + 16 + lcol, kk, half);

            v16bf b0 = ldsB(sb, 0, sub);
            v16bf b1 = ldsB(sb, 1, sub);
            v16bf b2 = ldsB(sb, 2, sub);
            air[0] = WMMA_BF16(ax0, b0, air[0]);  air[1] = WMMA_BF16(ax1, b0, air[1]);
            aiz[0] = WMMA_BF16(ax0, b1, aiz[0]);  aiz[1] = WMMA_BF16(ax1, b1, aiz[1]);
            ain[0] = WMMA_BF16(ax0, b2, ain[0]);  ain[1] = WMMA_BF16(ax1, b2, ain[1]);

            v16bf u0 = ldsB(sb, 3, sub);
            v16bf u1 = ldsB(sb, 4, sub);
            v16bf u2 = ldsB(sb, 5, sub);
            ahr[0] = WMMA_BF16(ah0, u0, ahr[0]);  ahr[1] = WMMA_BF16(ah1, u0, ahr[1]);
            ahz[0] = WMMA_BF16(ah0, u1, ahz[0]);  ahz[1] = WMMA_BF16(ah1, u1, ahz[1]);
            ahn[0] = WMMA_BF16(ah0, u2, ahn[0]);  ahn[1] = WMMA_BF16(ah1, u2, ahn[1]);
        }
    };

    stage(0, 0);
    stage_fence();

    int p = 0;
    for (int it = 0; it < HH / KSLAB - 1; ++it) {     // 15 steady-state slabs
        int k0 = it * KSLAB;
        stage(k0 + KSLAB, p ^ 1);
        // prefetch next activation slab (lowers to global_prefetch_b8)
        __builtin_prefetch(xb + (size_t)(m0 + lcol) * HH + k0 + 2 * KSLAB, 0, 1);
        __builtin_prefetch(hb + (size_t)(m0 + lcol) * HH + k0 + 2 * KSLAB, 0, 1);
        compute(k0, &smem[p * BUF_H]);
        stage_fence();
        p ^= 1;
    }
    compute(HH - KSLAB, &smem[p * BUF_H]);            // last slab

    // biases (per output column j, same for all C rows)
    const float br_  = bih[j] + bhh[j];
    const float bz_  = bih[HH + j] + bhh[HH + j];
    const float bin_ = bih[2 * HH + j];
    const float bhn_ = bhh[2 * HH + j];

#pragma unroll
    for (int tile = 0; tile < 2; ++tile) {
#pragma unroll
        for (int v = 0; v < 8; ++v) {
            int row = m0 + tile * 16 + v + half * 8;   // C layout: M = v + 8*laneHalf
            size_t o = (size_t)row * HH + j;
            float hold = hin[o];
            float r = sigm(air[tile][v] + ahr[tile][v] + br_);
            float z = sigm(aiz[tile][v] + ahz[tile][v] + bz_);
            float n = tanhf(ain[tile][v] + bin_ + r * (ahn[tile][v] + bhn_));
            float hnew = (1.0f - z) * n + z * hold;
            hout[o]  = hnew;
            xnext[o] = f2bf(hnew);
        }
    }
}

// logits = x @ dec_w.T + dec_b   (M=2048, N=256, K=1024)
__global__ void __launch_bounds__(256)
dec_gemm(const u16* __restrict__ xb, const u16* __restrict__ wb,
         const float* __restrict__ bias, float* __restrict__ out)
{
    const int lane = threadIdx.x & 31;
    const int wave = threadIdx.x >> 5;
    const int half = lane >> 4;
    const int lcol = lane & 15;
    const int n0 = blockIdx.x * 16;
    const int m0 = (blockIdx.y * 8 + wave) * 16;
    const int j  = n0 + lcol;

    v8f acc = {};
#pragma unroll 4
    for (int k0 = 0; k0 < HH; k0 += 32) {
        v16bf a = loadA(xb, m0 + lcol, k0, half);
        // direct-from-global B fragment (weight row j, K-contiguous)
        const u16* wp = wb + (size_t)j * HH + k0 + half * 8;
        FragU f; f.u[0] = *(const uint4*)(wp); f.u[1] = *(const uint4*)(wp + 16);
        acc = WMMA_BF16(a, f.v, acc);
    }
    float bj = bias[j];
#pragma unroll
    for (int v = 0; v < 8; ++v) {
        int row = m0 + v + half * 8;
        out[(size_t)row * OUTN + j] = acc[v] + bj;
    }
}

// ---------------- launcher ----------------
extern "C" void kernel_launch(void* const* d_in, const int* in_sizes, int n_in,
                              void* d_out, int out_size, void* d_ws, size_t ws_size,
                              hipStream_t stream) {
    const float* input  = (const float*)d_in[0];
    const float* hidden = (const float*)d_in[1];
    const float* i2h_w  = (const float*)d_in[2];
    const float* i2h_b  = (const float*)d_in[3];
    const float* w_ih   = (const float*)d_in[4];
    const float* w_hh   = (const float*)d_in[5];
    const float* b_ih   = (const float*)d_in[6];
    const float* b_hh   = (const float*)d_in[7];
    const float* dec_w  = (const float*)d_in[8];
    const float* dec_b  = (const float*)d_in[9];

    float* logits  = (float*)d_out;
    float* hid_out = logits + (size_t)BB * OUTN;

    // workspace layout (bf16 copies)
    const size_t W_EL = (size_t)LL * GG * HH;   // 12,582,912
    u16* wih_b  = (u16*)d_ws;
    u16* whh_b  = wih_b + W_EL;
    u16* h_b    = whh_b + W_EL;                 // L*B*H
    u16* decw_b = h_b   + (size_t)LL * BB * HH;
    u16* x_a    = decw_b + (size_t)OUTN * HH;
    u16* x_b    = x_a   + (size_t)BB * HH;

    const int T = 256;
    const int CVB = 4096;
    cvt_f32_bf16<<<CVB, T, 0, stream>>>(w_ih,  wih_b,  (int)W_EL);
    cvt_f32_bf16<<<CVB, T, 0, stream>>>(w_hh,  whh_b,  (int)W_EL);
    cvt_f32_bf16<<<CVB, T, 0, stream>>>(hidden, h_b,   LL * BB * HH);
    cvt_f32_bf16<<<CVB, T, 0, stream>>>(dec_w, decw_b, OUTN * HH);

    i2h_proj<<<(BB * HH + T - 1) / T, T, 0, stream>>>(input, i2h_w, i2h_b, x_a);

    // GRU layers: 64 N-tiles x 8 M-blocks (each block = 8 waves x 32 rows)
    dim3 ggrid(HH / 16, BB / (8 * 32));
    for (int l = 0; l < LL; ++l) {
        const u16* xin = (l & 1) ? x_b : x_a;
        u16* xout      = (l & 1) ? x_a : x_b;
        gru_layer<<<ggrid, T, 0, stream>>>(
            xin, h_b + (size_t)l * BB * HH,
            wih_b + (size_t)l * GG * HH, whh_b + (size_t)l * GG * HH,
            b_ih + (size_t)l * GG, b_hh + (size_t)l * GG,
            hidden + (size_t)l * BB * HH, hid_out + (size_t)l * BB * HH,
            xout);
    }

    // after layers 0..3 final activation is in x_a
    dim3 dgrid(OUTN / 16, BB / (16 * 8));
    dec_gemm<<<dgrid, T, 0, stream>>>(x_a, decw_b, dec_b, logits);
}
